// CrossAttention_52853867544576
// MI455X (gfx1250) — matrix operand order repaired and checked by reading, not compiled, not run on hardware
//
#include <hip/hip_runtime.h>

// ---------------------------------------------------------------------------
// CrossAttention for MI455X (gfx1250): bf16 WMMA GEMMs + softmax.
//   out = softmax( (T Wq^T)(S Wk^T)^T * D^-1/2 ) (S Wv^T) Wo^T
// GEMMs: NT, bf16 inputs, f32 WMMA accumulation (v_wmma_f32_16x16x32_bf16).
// Block tile 128x256 (8 waves, 2Mx4N; 64x64 per wave = 4x4 WMMA tiles).
// A/B tiles double-buffered in LDS via async global->LDS copies (ASYNCcnt),
// fragment reads are conflict-free ds_load_b128 (rows padded to 80B).
// ---------------------------------------------------------------------------

typedef __bf16 bf16_t;
typedef __attribute__((ext_vector_type(16))) __bf16 v16bf;
typedef __attribute__((ext_vector_type(8)))  __bf16 v8bf;
typedef __attribute__((ext_vector_type(8)))  float  v8f;

union V16U { v16bf v; v8bf h[2]; };

#define LDS_STRIDE 40                    // 32 data elems + 8 pad (80B rows)
#define OFF_A0 0                         // elems
#define OFF_A1 (128 * LDS_STRIDE)        // 5120
#define OFF_B0 (2 * 128 * LDS_STRIDE)    // 10240
#define OFF_B1 (OFF_B0 + 256 * LDS_STRIDE)
#define LDS_ELEMS (OFF_B1 + 256 * LDS_STRIDE)   // 30720 elems = 60KB

// Async copy of 16 bytes global -> LDS (GV mode, tracked by ASYNCcnt).
__device__ __forceinline__ void async_copy_b128(const bf16_t* gptr,
                                                unsigned lds_byte_off) {
    asm volatile("global_load_async_to_lds_b128 %0, %1, off"
                 :: "v"(lds_byte_off), "v"(gptr)
                 : "memory");
}

// A fragment from LDS tile (16x32 view, padded stride): lane L holds row
// m, elements 0..7 -> K=kb+0..7, 8..15 -> K=kb+16..23 (kb = 0 | 8).
__device__ __forceinline__ v16bf lds_a_frag(const bf16_t* tile, int m,
                                            int lane) {
    const int kb = (lane < 16) ? 0 : 8;
    const bf16_t* p = tile + m * LDS_STRIDE + kb;
    V16U u;
    u.h[0] = *(const v8bf*)(p);
    u.h[1] = *(const v8bf*)(p + 16);
    return u.v;
}

// B fragment from LDS tile: lane L holds col n, elems 0..7 -> K=kb+0..7,
// 8..15 -> K=kb+16..23 (kb = 0 | 8).
__device__ __forceinline__ v16bf lds_b_frag(const bf16_t* tile, int n,
                                            int lane) {
    const int kb = (lane < 16) ? 0 : 8;
    const bf16_t* p = tile + n * LDS_STRIDE + kb;
    V16U u;
    u.h[0] = *(const v8bf*)(p);
    u.h[1] = *(const v8bf*)(p + 16);
    return u.v;
}

// Issue async copies for one 32-wide K stage: A 128x32, B 256x32.
// 512 + 1024 16B-chunks over 256 threads = 2 + 4 = 6 asyncs per thread.
__device__ __forceinline__ void stage_async(const bf16_t* __restrict__ Ag,
                                            const bf16_t* __restrict__ Bg,
                                            int K, int kk,
                                            unsigned aoff, unsigned boff,
                                            int tid) {
#pragma unroll
    for (int i = 0; i < 2; ++i) {
        const int c = tid + i * 256;
        const int r = c >> 2, cc = c & 3;
        async_copy_b128(Ag + (size_t)r * K + kk + cc * 8,
                        aoff + (unsigned)(r * LDS_STRIDE + cc * 8) * 2u);
    }
#pragma unroll
    for (int i = 0; i < 4; ++i) {
        const int c = tid + i * 256;
        const int r = c >> 2, cc = c & 3;
        async_copy_b128(Bg + (size_t)r * K + kk + cc * 8,
                        boff + (unsigned)(r * LDS_STRIDE + cc * 8) * 2u);
    }
}

// C[m][n] = alpha * sum_k A[m][k] * B[n][k]   (leading dims == K everywhere)
template <bool F32OUT>
__global__ __launch_bounds__(256) void gemm_nt_bf16(
    const bf16_t* __restrict__ A, const bf16_t* __restrict__ Bm,
    void* __restrict__ Cv, int M, int N, int K,
    long sA, long sB, long sC, float alpha)
{
    __shared__ alignas(64) bf16_t lds[LDS_ELEMS];

    const int tid   = threadIdx.x;
    const int lane  = tid & 31;
    const int wave  = tid >> 5;
    const int waveM = wave & 1;
    const int waveN = wave >> 1;
    const int lm    = lane & 15;
    const long z    = blockIdx.z;

    const bf16_t* Ag = A  + z * sA + (size_t)(blockIdx.y * 128) * K;
    const bf16_t* Bg = Bm + z * sB + (size_t)(blockIdx.x * 256) * K;

    const unsigned aoff[2] = { (unsigned)(uintptr_t)&lds[OFF_A0],
                               (unsigned)(uintptr_t)&lds[OFF_A1] };
    const unsigned boff[2] = { (unsigned)(uintptr_t)&lds[OFF_B0],
                               (unsigned)(uintptr_t)&lds[OFF_B1] };

    v8f zero = {};
    v8f acc[4][4];
#pragma unroll
    for (int i = 0; i < 4; ++i)
#pragma unroll
        for (int j = 0; j < 4; ++j)
            acc[i][j] = zero;

    const int nk = K / 32;
    stage_async(Ag, Bg, K, 0, aoff[0], boff[0], tid);

    for (int k = 0; k < nk; ++k) {
        if (k + 1 < nk) {
            stage_async(Ag, Bg, K, (k + 1) * 32,
                        aoff[(k + 1) & 1], boff[(k + 1) & 1], tid);
            // stage k done when only the 6 just-issued remain (in-order)
            asm volatile("s_wait_asynccnt 0x6" ::: "memory");
        } else {
            asm volatile("s_wait_asynccnt 0x0" ::: "memory");
        }
        __syncthreads();   // all waves' share of stage k visible

        const bf16_t* At = lds + ((k & 1) ? OFF_A1 : OFF_A0);
        const bf16_t* Bt = lds + ((k & 1) ? OFF_B1 : OFF_B0);

        v16bf af[4], bfr[4];
#pragma unroll
        for (int i = 0; i < 4; ++i)
            af[i] = lds_a_frag(At, waveM * 64 + i * 16 + lm, lane);
#pragma unroll
        for (int j = 0; j < 4; ++j)
            bfr[j] = lds_b_frag(Bt, waveN * 64 + j * 16 + lm, lane);
#pragma unroll
        for (int i = 0; i < 4; ++i)
#pragma unroll
            for (int j = 0; j < 4; ++j)
                acc[i][j] = __builtin_amdgcn_wmma_f32_16x16x32_bf16(
                    false, af[i], false, bfr[j], (short)0, acc[i][j],
                    false, false);

        __syncthreads();   // done reading buf[k&1] before it is re-filled
    }

    // C layout: lane holds col n=lm, rows rbase+0..7 (rbase=8 for lanes>=16)
    const int mbase = blockIdx.y * 128 + waveM * 64;
    const int nbase = blockIdx.x * 256 + waveN * 64;
    const int rbase = (lane < 16) ? 0 : 8;
#pragma unroll
    for (int i = 0; i < 4; ++i) {
#pragma unroll
        for (int j = 0; j < 4; ++j) {
            const int col = nbase + j * 16 + lm;
#pragma unroll
            for (int r = 0; r < 8; ++r) {
                const size_t row = (size_t)(mbase + i * 16 + rbase + r);
                const float  val = acc[i][j][r] * alpha;
                if (F32OUT)
                    ((float*)Cv)[(size_t)z * sC + row * N + col] = val;
                else
                    ((bf16_t*)Cv)[(size_t)z * sC + row * N + col] = (bf16_t)val;
            }
        }
    }
}

__global__ void cvt_f32_bf16(const float* __restrict__ in,
                             bf16_t* __restrict__ out, size_t n) {
    size_t i = (size_t)blockIdx.x * blockDim.x + threadIdx.x;
    const size_t stride = (size_t)gridDim.x * blockDim.x;
    for (; i < n; i += stride) out[i] = (bf16_t)in[i];
}

// In-place row softmax over bf16, one 256-thread block per row, f32 math.
__global__ __launch_bounds__(256) void softmax_rows(bf16_t* __restrict__ P,
                                                    int ncols) {
    bf16_t* p = P + (size_t)blockIdx.x * ncols;
    const int t = threadIdx.x;
    const int per = ncols / 256;   // 16 for ncols=4096
    float ex[16];
    float m = -3.0e38f;
    for (int i = 0; i < per; ++i) {
        float v = (float)p[t + i * 256];
        ex[i] = v;
        m = fmaxf(m, v);
    }
    __shared__ float red[256];
    red[t] = m; __syncthreads();
    for (int s = 128; s > 0; s >>= 1) {
        if (t < s) red[t] = fmaxf(red[t], red[t + s]);
        __syncthreads();
    }
    m = red[0]; __syncthreads();
    float sum = 0.f;
    for (int i = 0; i < per; ++i) { ex[i] = __expf(ex[i] - m); sum += ex[i]; }
    red[t] = sum; __syncthreads();
    for (int s = 128; s > 0; s >>= 1) {
        if (t < s) red[t] += red[t + s];
        __syncthreads();
    }
    const float inv = 1.0f / red[0];
    for (int i = 0; i < per; ++i) p[t + i * 256] = (bf16_t)(ex[i] * inv);
}

// out[c][r] = in[r][c] per batch z; 32x32 LDS tile, 256 threads.
__global__ __launch_bounds__(256) void transpose_bf16(
    const bf16_t* __restrict__ in, bf16_t* __restrict__ out,
    int rows, int cols) {
    __shared__ bf16_t tile[32][33];
    const size_t zoff = (size_t)blockIdx.z * rows * cols;
    const bf16_t* inz  = in  + zoff;
    bf16_t*       outz = out + zoff;
    const int c0 = blockIdx.x * 32, r0 = blockIdx.y * 32;
    const int tx = threadIdx.x & 31, ty = threadIdx.x >> 5;   // 32 x 8
#pragma unroll
    for (int i = 0; i < 4; ++i) {
        const int r = r0 + ty + i * 8;
        tile[ty + i * 8][tx] = inz[(size_t)r * cols + c0 + tx];
    }
    __syncthreads();
#pragma unroll
    for (int i = 0; i < 4; ++i) {
        const int c = c0 + ty + i * 8;
        outz[(size_t)c * rows + r0 + tx] = tile[tx][ty + i * 8];
    }
}

extern "C" void kernel_launch(void* const* d_in, const int* in_sizes, int n_in,
                              void* d_out, int out_size, void* d_ws,
                              size_t ws_size, hipStream_t stream) {
    constexpr int B = 8, LQ = 1024, LK = 4096, D = 1024;
    const float* T  = (const float*)d_in[0];
    const float* S  = (const float*)d_in[1];
    const float* Wq = (const float*)d_in[2];
    const float* Wk = (const float*)d_in[3];
    const float* Wv = (const float*)d_in[4];
    const float* Wo = (const float*)d_in[5];

    char* ws = (char*)d_ws;
    const size_t MB = 1024ull * 1024ull;
    // Workspace layout (bytes), total 296 MB:
    bf16_t* Tbf = (bf16_t*)(ws + 0);          //  16 MB: T bf16   (8M elems)
    bf16_t* Sbf = (bf16_t*)(ws + 16 * MB);    //  64 MB: S bf16   (32M elems)
    bf16_t* Wqb = (bf16_t*)(ws + 80 * MB);    //   2 MB each: weights bf16
    bf16_t* Wkb = (bf16_t*)(ws + 82 * MB);
    bf16_t* Wvb = (bf16_t*)(ws + 84 * MB);
    bf16_t* Wob = (bf16_t*)(ws + 86 * MB);
    bf16_t* Qb  = (bf16_t*)(ws + 88 * MB);    //  16 MB: Q
    bf16_t* Kb  = (bf16_t*)(ws + 104 * MB);   //  64 MB: K
    bf16_t* Vb  = (bf16_t*)(ws + 168 * MB);   //  64 MB: V
    bf16_t* Scr = (bf16_t*)(ws + 232 * MB);   //  64 MB: scores/P
    bf16_t* Vt  = Sbf;   // reuse S buffer for V^T after K,V projections
    bf16_t* Ob  = Tbf;   // reuse T buffer for attention output after Q proj

    // 1) fp32 -> bf16 conversions
    cvt_f32_bf16<<<4096, 256, 0, stream>>>(T,  Tbf, (size_t)B * LQ * D);
    cvt_f32_bf16<<<4096, 256, 0, stream>>>(S,  Sbf, (size_t)B * LK * D);
    cvt_f32_bf16<<<1024, 256, 0, stream>>>(Wq, Wqb, (size_t)D * D);
    cvt_f32_bf16<<<1024, 256, 0, stream>>>(Wk, Wkb, (size_t)D * D);
    cvt_f32_bf16<<<1024, 256, 0, stream>>>(Wv, Wvb, (size_t)D * D);
    cvt_f32_bf16<<<1024, 256, 0, stream>>>(Wo, Wob, (size_t)D * D);

    // 2) Projections (batch folded into M; weights shared)
    gemm_nt_bf16<false><<<dim3(D / 256, (B * LQ) / 128, 1), 256, 0, stream>>>(
        Tbf, Wqb, Qb, B * LQ, D, D, 0, 0, 0, 1.0f);
    gemm_nt_bf16<false><<<dim3(D / 256, (B * LK) / 128, 1), 256, 0, stream>>>(
        Sbf, Wkb, Kb, B * LK, D, D, 0, 0, 0, 1.0f);
    gemm_nt_bf16<false><<<dim3(D / 256, (B * LK) / 128, 1), 256, 0, stream>>>(
        Sbf, Wvb, Vb, B * LK, D, D, 0, 0, 0, 1.0f);

    // 3) V^T per batch (LK x D -> D x LK) so PV is an NT GEMM
    transpose_bf16<<<dim3(D / 32, LK / 32, B), 256, 0, stream>>>(Vb, Vt, LK, D);

    // 4) scores = Q K^T * D^-1/2  (per batch)
    gemm_nt_bf16<false><<<dim3(LK / 256, LQ / 128, B), 256, 0, stream>>>(
        Qb, Kb, Scr, LQ, LK, D,
        (long)LQ * D, (long)LK * D, (long)LQ * LK, 0.03125f);

    // 5) softmax rows (in place)
    softmax_rows<<<B * LQ, 256, 0, stream>>>(Scr, LK);

    // 6) O = P V   (NT against V^T, per batch)
    gemm_nt_bf16<false><<<dim3(D / 256, LQ / 128, B), 256, 0, stream>>>(
        Scr, Vt, Ob, LQ, D, LK,
        (long)LQ * LK, (long)D * LK, (long)LQ * D, 1.0f);

    // 7) out = O Wo^T  (f32 output to d_out)
    gemm_nt_bf16<true><<<dim3(D / 256, (B * LQ) / 128, 1), 256, 0, stream>>>(
        Ob, Wob, d_out, B * LQ, D, D, 0, 0, 0, 1.0f);
}
